// ResidualBlock_78761110274311
// MI455X (gfx1250) — matrix operand order repaired and checked by reading, not compiled, not run on hardware
//
#include <hip/hip_runtime.h>
#include <hip/hip_fp16.h>

typedef __attribute__((ext_vector_type(16))) _Float16 v16h;
typedef __attribute__((ext_vector_type(8)))  _Float16 v8h;
typedef __attribute__((ext_vector_type(8)))  float    v8f;

#define NB      4
#define LSEQ    2048
#define DMODEL  1024
#define DINNER  2048
#define DSTATE  32
#define DCONV   16
#define DTRANK  64
#define NTOK    (NB * LSEQ)        // 8192 tokens

// ---------------------------------------------------------------------------
// f32 -> f16 conversion (weights / activations)
// ---------------------------------------------------------------------------
__global__ __launch_bounds__(256) void f32_to_f16(const float* __restrict__ s,
                                                  _Float16* __restrict__ d, int n) {
  int i = blockIdx.x * 256 + threadIdx.x;
  if (i < n) d[i] = (_Float16)s[i];
}

// ---------------------------------------------------------------------------
// RMSNorm over d_model per token, emit f16 for the in_proj WMMA GEMM
// ---------------------------------------------------------------------------
__global__ __launch_bounds__(256) void rmsnorm_f16(const float* __restrict__ x,
                                                   const float* __restrict__ w,
                                                   _Float16* __restrict__ o) {
  int tok = blockIdx.x;
  const float* xr = x + (size_t)tok * DMODEL;
  float ss = 0.f;
  for (int i = threadIdx.x; i < DMODEL; i += 256) { float v = xr[i]; ss += v * v; }
  for (int m = 16; m >= 1; m >>= 1) ss += __shfl_xor(ss, m);
  __shared__ float red[8];
  __shared__ float stot;
  if ((threadIdx.x & 31) == 0) red[threadIdx.x >> 5] = ss;
  __syncthreads();
  if (threadIdx.x == 0) { float t = 0.f; for (int i = 0; i < 8; ++i) t += red[i]; stot = t; }
  __syncthreads();
  float inv = rsqrtf(stot / (float)DMODEL + 1e-5f);
  for (int i = threadIdx.x; i < DMODEL; i += 256)
    o[(size_t)tok * DMODEL + i] = (_Float16)(xr[i] * inv * w[i]);
}

// ---------------------------------------------------------------------------
// WMMA f16->f32 GEMM:  C[M,N] = A[M,K] * W[N,K]^T  (+ fused epilogue)
// block = 256 threads = 8 waves; wave tile = 32(M) x 64(N); block tile 256x64.
// B tile (64 N-rows x 64 K-halfs = 8KB) is staged to LDS with
// GLOBAL_LOAD_ASYNC_TO_LDS_B128 (ASYNCcnt), double-buffered; steady-state loop
// unconditionally issues stage k+1 then s_wait_asynccnt 2; last stage peeled.
// Each ds_load_b128-pair B fragment feeds TWO WMMAs (2 M-subtiles per wave).
// Fragment layouts used here (wave32):
//   A: lane<16 row m holds K 0..15; lane>=16 same row, K 16..31
//   B: lane holds column n (= row n of W), K = half*16 + {0..15}  (32B contig)
//   C/D: VGPR r -> M = r + half*8 ; N = lane&15
// EPI: 0 = plain f32 store, 1 = softplus(acc + bias[n]), 2 = acc + res[m,n]
// ---------------------------------------------------------------------------
template <int EPI>
__global__ __launch_bounds__(256)
void gemm_f16_wmma(const _Float16* __restrict__ A, const _Float16* __restrict__ W,
                   float* __restrict__ C, int M, int N, int K,
                   const float* __restrict__ bias, const float* __restrict__ res) {
  __shared__ _Float16 sB[2][64 * 64];          // [buf][n(64) x k(64)]  2 x 8KB
  const int tid  = threadIdx.x;
  const int lane = tid & 31;
  const int wave = tid >> 5;
  const int half = lane >> 4;
  const int l16  = lane & 15;
  const int m0 = blockIdx.x * 256 + wave * 32;
  const int n0 = blockIdx.y * 64;

  // async-copy addressing: 4 threads per B row, 32B per thread (2 x b128)
  const int crow = tid >> 2;
  const int cseg = tid & 3;
  const uint64_t wbase = (uint64_t)(uintptr_t)W;
  const uint32_t goff0 = (uint32_t)((((size_t)(n0 + crow) * K) + (size_t)cseg * 16)
                                    * sizeof(_Float16));
  const uint32_t lds0  = (uint32_t)(uintptr_t)(&sB[0][0]) + crow * 128 + cseg * 32;

  auto issueB = [&](int buf, int kb) {
    uint32_t la = lds0 + (uint32_t)buf * (64 * 64 * 2);
    uint32_t ga = goff0 + (uint32_t)(kb * 2);
    asm volatile("global_load_async_to_lds_b128 %0, %1, %2\n\t"
                 "global_load_async_to_lds_b128 %0, %1, %2 offset:16"
                 :: "v"(la), "v"(ga), "s"(wbase) : "memory");
  };

  auto loadA = [&](const _Float16* p) -> v16h {
    v8h lo = *(const v8h*)p;                              // K  0..7 / 16..23
    v8h hi = *(const v8h*)(p + 8);                        // K  8..15 / 24..31
    return __builtin_shufflevector(lo, hi, 0, 1, 2, 3, 4, 5, 6, 7,
                                   8, 9, 10, 11, 12, 13, 14, 15);
  };

  v8f acc0[4] = {};                                       // M rows m0   .. m0+15
  v8f acc1[4] = {};                                       // M rows m0+16.. m0+31
  const _Float16* arow0 = A + (size_t)(m0 + l16) * K + half * 16;
  const _Float16* arow1 = arow0 + (size_t)16 * K;

  auto stage = [&](int bufc, int kb) {
#pragma unroll
    for (int kk = 0; kk < 64; kk += 32) {
      v16h a0 = loadA(arow0 + kb + kk);
      v16h a1 = loadA(arow1 + kb + kk);
      const _Float16* bbase = &sB[bufc][kk + half * 16];
#pragma unroll
      for (int t = 0; t < 4; ++t) {
        v16h b = *(const v16h*)(bbase + (size_t)(16 * t + l16) * 64);  // ds_load
        acc0[t] = __builtin_amdgcn_wmma_f32_16x16x32_f16(
            false, a0, false, b, (short)0, acc0[t], false, false);
        acc1[t] = __builtin_amdgcn_wmma_f32_16x16x32_f16(
            false, a1, false, b, (short)0, acc1[t], false, false);
      }
    }
  };

  issueB(0, 0);
  int buf = 0;
  // steady state: always a next stage -> branch-free wait on asynccnt<=2
  for (int kb = 0; kb + 64 < K; kb += 64) {
    issueB(buf ^ 1, kb + 64);
    asm volatile("s_wait_asynccnt 0x2" ::: "memory");     // stage kb landed
    __syncthreads();                                      // publish B tile
    __builtin_prefetch(arow0 + kb + 128, 0, 1);           // global_prefetch_b8
    stage(buf, kb);
    __syncthreads();                                      // done reading buf
    buf ^= 1;
  }
  // peeled last stage
  asm volatile("s_wait_asynccnt 0x0" ::: "memory");
  __syncthreads();
  stage(buf, K - 64);

#pragma unroll
  for (int g = 0; g < 2; ++g) {
    const v8f* accg = g ? acc1 : acc0;
#pragma unroll
    for (int t = 0; t < 4; ++t) {
      int cn = n0 + 16 * t + l16;
#pragma unroll
      for (int r = 0; r < 8; ++r) {
        int cm = m0 + g * 16 + half * 8 + r;
        float v = accg[t][r];
        if (EPI == 1) {                                   // softplus(acc + bias)
          v += bias[cn];
          v = (v > 20.f) ? v : __logf(1.f + __expf(v));
        }
        if (EPI == 2) v += res[(size_t)cm * N + cn];      // residual add
        C[(size_t)cm * N + cn] = v;
      }
    }
  }
}

// ---------------------------------------------------------------------------
// Causal depthwise conv (taps=16) + bias + SiLU.  xi = xz[:, :DINNER].
// Writes f32 copy (scan / D*xc path) and f16 copy (x_proj GEMM operand).
// ---------------------------------------------------------------------------
__global__ __launch_bounds__(256)
void conv_silu(const float* __restrict__ xz, const float* __restrict__ cw,
               const float* __restrict__ cb, float* __restrict__ xc,
               _Float16* __restrict__ xch) {
  int tok = blockIdx.x;            // b*L + l
  int l = tok & (LSEQ - 1);
  for (int d = threadIdx.x; d < DINNER; d += 256) {
    float wv[DCONV];
#pragma unroll
    for (int j = 0; j < DCONV; ++j) wv[j] = cw[d * DCONV + j];
    float acc = cb[d];
#pragma unroll
    for (int j = 0; j < DCONV; ++j) {
      int t = l - (DCONV - 1) + j;
      if (t >= 0)
        acc += xz[(size_t)(tok - (DCONV - 1) + j) * (2 * DINNER) + d] * wv[j];
    }
    float sv = acc / (1.f + __expf(-acc));              // SiLU
    size_t o = (size_t)tok * DINNER + d;
    xc[o] = sv;
    xch[o] = (_Float16)sv;
  }
}

// ---------------------------------------------------------------------------
// Pull dt low-rank slice out of x_dbl (f32 [NTOK,128]) as f16 [NTOK,64]
// ---------------------------------------------------------------------------
__global__ __launch_bounds__(256)
void extract_dtlow(const float* __restrict__ xdbl, _Float16* __restrict__ d) {
  int i = blockIdx.x * 256 + threadIdx.x;               // < NTOK*DTRANK
  int m = i >> 6, r = i & 63;
  d[i] = (_Float16)xdbl[(size_t)m * 128 + r];
}

// ---------------------------------------------------------------------------
// Selective scan: one wave32 per (b,d) channel; lane = state s (DSTATE==32).
// h_s = exp(dt*A_s)*h_s + dt*x*B_s ;  y_t = wave_reduce(h_s * C_s)
// ---------------------------------------------------------------------------
__global__ __launch_bounds__(256)
void scan_kernel(const float* __restrict__ dt, const float* __restrict__ xc,
                 const float* __restrict__ xdbl, const float* __restrict__ Alog,
                 float* __restrict__ ypre) {
  int wg = blockIdx.x * 8 + (threadIdx.x >> 5);          // 0..8191
  int lane = threadIdx.x & 31;
  int b = wg >> 11;                                      // / DINNER
  int d = wg & (DINNER - 1);
  float As = -__expf(Alog[(size_t)d * DSTATE + lane]);
  const float* dtp = dt + (size_t)b * LSEQ * DINNER + d;
  const float* xp  = xc + (size_t)b * LSEQ * DINNER + d;
  const float* Bp  = xdbl + (size_t)b * LSEQ * 128 + DTRANK + lane;
  const float* Cp  = Bp + DSTATE;
  float* yp = ypre + (size_t)b * LSEQ * DINNER + d;
  float h = 0.f;
  for (int t = 0; t < LSEQ; ++t) {
    float dtv = dtp[(size_t)t * DINNER];                 // wave-uniform
    float xv  = xp[(size_t)t * DINNER];
    float Bv  = Bp[(size_t)t * 128];
    float Cv  = Cp[(size_t)t * 128];
    h = __expf(dtv * As) * h + (dtv * xv) * Bv;
    float v = h * Cv;
    v += __shfl_xor(v, 16); v += __shfl_xor(v, 8); v += __shfl_xor(v, 4);
    v += __shfl_xor(v, 2);  v += __shfl_xor(v, 1);
    if (lane == 0) yp[(size_t)t * DINNER] = v;
  }
}

// ---------------------------------------------------------------------------
// y = (y_scan + D*xc) * silu(z) -> f16 operand for out_proj GEMM
// ---------------------------------------------------------------------------
__global__ __launch_bounds__(256)
void gate_kernel(const float* __restrict__ ypre, const float* __restrict__ xc,
                 const float* __restrict__ Dp, const float* __restrict__ xz,
                 _Float16* __restrict__ yh) {
  size_t i = (size_t)blockIdx.x * 256 + threadIdx.x;     // < NTOK*DINNER
  int d = (int)(i & (DINNER - 1));
  size_t m = i >> 11;
  float z = xz[m * (2 * DINNER) + DINNER + d];
  float g = z / (1.f + __expf(-z));
  yh[i] = (_Float16)((ypre[i] + Dp[d] * xc[i]) * g);
}

// ---------------------------------------------------------------------------
extern "C" void kernel_launch(void* const* d_in, const int* in_sizes, int n_in,
                              void* d_out, int out_size, void* d_ws, size_t ws_size,
                              hipStream_t stream) {
  const float* x          = (const float*)d_in[0];
  const float* norm_w     = (const float*)d_in[1];
  const float* in_proj_w  = (const float*)d_in[2];   // [4096,1024]
  const float* conv_w     = (const float*)d_in[3];   // [2048,16]
  const float* conv_b     = (const float*)d_in[4];
  const float* x_proj_w   = (const float*)d_in[5];   // [128,2048]
  const float* dt_proj_w  = (const float*)d_in[6];   // [2048,64]
  const float* dt_proj_b  = (const float*)d_in[7];
  const float* A_log      = (const float*)d_in[8];
  const float* D_param    = (const float*)d_in[9];
  const float* out_proj_w = (const float*)d_in[10];  // [1024,2048]
  float* out = (float*)d_out;

  char* ws = (char*)d_ws;
  size_t off = 0;
  auto alloc = [&](size_t bytes) {
    void* p = ws + off;
    off = (off + bytes + 255) & ~(size_t)255;
    return p;
  };

  _Float16* xn_h    = (_Float16*)alloc((size_t)NTOK * DMODEL * 2);
  _Float16* w_in_h  = (_Float16*)alloc((size_t)2 * DINNER * DMODEL * 2);
  _Float16* w_xp_h  = (_Float16*)alloc((size_t)(DTRANK + 2 * DSTATE) * DINNER * 2);
  _Float16* w_dt_h  = (_Float16*)alloc((size_t)DINNER * DTRANK * 2);
  _Float16* w_out_h = (_Float16*)alloc((size_t)DMODEL * DINNER * 2);
  _Float16* dtlow_h = (_Float16*)alloc((size_t)NTOK * DTRANK * 2);
  float*    xz      = (float*)alloc((size_t)NTOK * 2 * DINNER * 4);
  float*    xc      = (float*)alloc((size_t)NTOK * DINNER * 4);
  _Float16* xc_h    = (_Float16*)alloc((size_t)NTOK * DINNER * 2);
  float*    x_dbl   = (float*)alloc((size_t)NTOK * 128 * 4);
  float*    dtf     = (float*)alloc((size_t)NTOK * DINNER * 4);
  float*    y_pre   = (float*)alloc((size_t)NTOK * DINNER * 4);
  _Float16* y_h     = (_Float16*)alloc((size_t)NTOK * DINNER * 2);

  // 1. weight conversions (f32 -> f16)
  f32_to_f16<<<(2 * DINNER * DMODEL + 255) / 256, 256, 0, stream>>>(in_proj_w, w_in_h, 2 * DINNER * DMODEL);
  f32_to_f16<<<(128 * DINNER + 255) / 256, 256, 0, stream>>>(x_proj_w, w_xp_h, 128 * DINNER);
  f32_to_f16<<<(DINNER * DTRANK + 255) / 256, 256, 0, stream>>>(dt_proj_w, w_dt_h, DINNER * DTRANK);
  f32_to_f16<<<(DMODEL * DINNER + 255) / 256, 256, 0, stream>>>(out_proj_w, w_out_h, DMODEL * DINNER);

  // 2. RMSNorm -> f16
  rmsnorm_f16<<<NTOK, 256, 0, stream>>>(x, norm_w, xn_h);

  // 3. in_proj: [8192,1024] x [4096,1024]^T -> xz f32
  gemm_f16_wmma<0><<<dim3(NTOK / 256, (2 * DINNER) / 64), 256, 0, stream>>>(
      xn_h, w_in_h, xz, NTOK, 2 * DINNER, DMODEL, nullptr, nullptr);

  // 4. depthwise causal conv + SiLU
  conv_silu<<<NTOK, 256, 0, stream>>>(xz, conv_w, conv_b, xc, xc_h);

  // 5. x_proj: [8192,2048] x [128,2048]^T -> x_dbl f32
  gemm_f16_wmma<0><<<dim3(NTOK / 256, 128 / 64), 256, 0, stream>>>(
      xc_h, w_xp_h, x_dbl, NTOK, 128, DINNER, nullptr, nullptr);

  // 6. dt low-rank slice -> f16
  extract_dtlow<<<(NTOK * DTRANK) / 256, 256, 0, stream>>>(x_dbl, dtlow_h);

  // 7. dt_proj + bias + softplus: [8192,64] x [2048,64]^T -> dt f32
  gemm_f16_wmma<1><<<dim3(NTOK / 256, DINNER / 64), 256, 0, stream>>>(
      dtlow_h, w_dt_h, dtf, NTOK, DINNER, DTRANK, dt_proj_b, nullptr);

  // 8. selective scan (wave per channel, lane per state)
  scan_kernel<<<(NB * DINNER) / 8, 256, 0, stream>>>(dtf, xc, x_dbl, A_log, y_pre);

  // 9. gating: (y + D*xc) * silu(z) -> f16
  gate_kernel<<<(NTOK * DINNER) / 256, 256, 0, stream>>>(y_pre, xc, D_param, xz, y_h);

  // 10. out_proj + residual: [8192,2048] x [1024,2048]^T + x -> out
  gemm_f16_wmma<2><<<dim3(NTOK / 256, DMODEL / 64), 256, 0, stream>>>(
      y_h, w_out_h, out, NTOK, DMODEL, DINNER, nullptr, x);
}
